// IpexWoqLinear_11751030522137
// MI455X (gfx1250) — compile-verified, matrix operand order checked by reading, not run on hardware
//
#include <hip/hip_runtime.h>
#include <hip/hip_bf16.h>

typedef __attribute__((ext_vector_type(16))) _Float16     v16h;
typedef __attribute__((ext_vector_type(8)))  _Float16     v8h;
typedef __attribute__((ext_vector_type(8)))  float        v8f;
typedef __attribute__((ext_vector_type(8)))  unsigned int v8u;
typedef __attribute__((ext_vector_type(4)))  unsigned int v4u;

#define IN_F  8192
#define OUT_F 8192
#define TOK   64

// ---------------------------------------------------------------------------
// Kernel 1: x fp32 -> f16, plus per-token row sums of the f16-rounded x for
// the (zero-point + 1024) correction term.
// ws layout: [0,1MB) = xh f16, then 64 floats of rowsum.
// ---------------------------------------------------------------------------
__global__ __launch_bounds__(256) void woq_prep(const float* __restrict__ x,
                                                _Float16* __restrict__ xh,
                                                float* __restrict__ rowsum) {
  __shared__ float red[256];
  const int row = blockIdx.x;
  const int tid = threadIdx.x;
  const float* xr = x  + (size_t)row * IN_F;
  _Float16*    xo = xh + (size_t)row * IN_F;
  float s = 0.f;
  for (int k = tid; k < IN_F; k += 256) {
    _Float16 h = (_Float16)xr[k];
    xo[k] = h;
    s += (float)h;
  }
  red[tid] = s;
  __syncthreads();
  for (int off = 128; off > 0; off >>= 1) {
    if (tid < off) red[tid] += red[tid + off];
    __syncthreads();
  }
  if (tid == 0) rowsum[row] = red[0];
}

// ---------------------------------------------------------------------------
// Kernel 2: f16 WMMA GEMM of x[64xK] against biased integer codes.
// Codes q in [0,255] are materialized as f16 (1024+q) with pure bit ops:
//   as_f16(0x6400 | q) == 1024 + q   (exact; f16 holds ints <= 2048 exactly)
// The +1024 offset folds into the zero-point term in the epilogue:
//   y[t,o] = s[o] * (acc[t,o] - (zp[o]+1024)*rowsum[t]) + bias[o]
// Block = 128 threads = 4 waves; wave w owns token tile [16w,16w+16) x the
// block's 16 output columns, full K.  All 4 waves stream the SAME qweight
// cachelines -> HBM sees qweight exactly once; x (1MB f16) stays in L2.
// ---------------------------------------------------------------------------
__global__ __launch_bounds__(128) void woq_gemm(
    const _Float16* __restrict__ xh, const unsigned int* __restrict__ qw,
    const float* __restrict__ scales, const int* __restrict__ zps,
    const float* __restrict__ bias, const float* __restrict__ rowsum,
    float* __restrict__ out) {
  const int lane = threadIdx.x & 31;
  const int wave = threadIdx.x >> 5;   // 0..3 -> token (M) tile
  const int m0   = wave << 4;
  const int o0   = blockIdx.x << 4;
  const int half = lane >> 4;          // K half-wave striping
  const int lm   = lane & 15;

  // A (16x32 f16): lane holds row m0+lm, K = half*8 + {0..7} and +16.
  const _Float16* aptr = xh + (size_t)(m0 + lm) * IN_F + (half << 3);
  // B (32x16): lane holds output column o0+lm == qweight row (contiguous),
  // same K striping as A.
  const unsigned int* bptr = qw + (size_t)(o0 + lm) * IN_F + (half << 3);

  v8f c = {};
#pragma unroll 4
  for (int k0 = 0; k0 < IN_F; k0 += 32) {
    // ---- A tile ----
    v8h alo = *(const v8h*)(aptr);        // K = half*8 + 0..7
    v8h ahi = *(const v8h*)(aptr + 16);   // K = 16 + half*8 + 0..7
    v16h a = __builtin_shufflevector(alo, ahi,
                                     0,1,2,3,4,5,6,7,8,9,10,11,12,13,14,15);

    // ---- B tile: 16 int32 codes -> f16 (1024+q) via bit ops ----
    v4u q0 = *(const v4u*)(bptr + 0);     // K = half*8 + 0..3
    v4u q1 = *(const v4u*)(bptr + 4);     // K = half*8 + 4..7
    v4u q2 = *(const v4u*)(bptr + 16);    // K = 16 + half*8 + 0..3
    v4u q3 = *(const v4u*)(bptr + 20);    // K = 16 + half*8 + 4..7
    v8u bu;
    bu[0] = (q0[0] | (q0[1] << 16)) | 0x64006400u;
    bu[1] = (q0[2] | (q0[3] << 16)) | 0x64006400u;
    bu[2] = (q1[0] | (q1[1] << 16)) | 0x64006400u;
    bu[3] = (q1[2] | (q1[3] << 16)) | 0x64006400u;
    bu[4] = (q2[0] | (q2[1] << 16)) | 0x64006400u;
    bu[5] = (q2[2] | (q2[3] << 16)) | 0x64006400u;
    bu[6] = (q3[0] | (q3[1] << 16)) | 0x64006400u;
    bu[7] = (q3[2] | (q3[3] << 16)) | 0x64006400u;
    v16h b = __builtin_bit_cast(v16h, bu);

    c = __builtin_amdgcn_wmma_f32_16x16x32_f16(
        /*neg_a=*/false, a, /*neg_b=*/false, b,
        /*c_mod=*/(short)0, c, /*reuse_a=*/false, /*reuse_b=*/false);

    aptr += 32;
    bptr += 32;
    // Keep the qweight HBM stream ahead (~2KB / 16 iterations per lane).
    // locality=3 -> near-scope (WGP) prefetch: pulls into all cache levels,
    // so the 4 waves sharing these lines hit in WGP$ instead of GL2.
    __builtin_prefetch(bptr + 16 * 32, 0, 3);
  }

  // ---- Epilogue: per-output-channel affine + bias (+1024 bias fold) ----
  const int   o   = o0 + lm;
  const float s   = scales[o];
  const float zpf = (float)zps[o] + 1024.0f;
  const float bb  = bias[o];
  const int   t0  = m0 + (half << 3);   // D: VGPR r <-> row r + 8*(lane/16)
#pragma unroll
  for (int r = 0; r < 8; ++r) {
    const int t = t0 + r;
    out[(size_t)t * OUT_F + o] = s * (c[r] - zpf * rowsum[t]) + bb;
  }
}

extern "C" void kernel_launch(void* const* d_in, const int* in_sizes, int n_in,
                              void* d_out, int out_size, void* d_ws, size_t ws_size,
                              hipStream_t stream) {
  const float*        x      = (const float*)d_in[0];
  const unsigned int* qw     = (const unsigned int*)d_in[1];
  const float*        scales = (const float*)d_in[2];
  const int*          zps    = (const int*)d_in[3];
  const float*        bias   = (const float*)d_in[4];
  float*              out    = (float*)d_out;

  _Float16* xh     = (_Float16*)d_ws;                                   // 1 MB
  float*    rowsum = (float*)((char*)d_ws +
                              (size_t)TOK * IN_F * sizeof(_Float16));   // 256 B

  woq_prep<<<TOK, 256, 0, stream>>>(x, xh, rowsum);
  woq_gemm<<<OUT_F / 16, 128, 0, stream>>>(xh, qw, scales, zps, bias, rowsum, out);
}
